// GPRGNN_66005057405289
// MI455X (gfx1250) — matrix operand order repaired and checked by reading, not compiled, not run on hardware
//
#include <hip/hip_runtime.h>
#include <stdint.h>

typedef __bf16 bf16_t;
typedef __attribute__((ext_vector_type(16))) __bf16 v16bf;
typedef __attribute__((ext_vector_type(8)))  __bf16 v8bf;
typedef __attribute__((ext_vector_type(8)))  float  v8f;

#define NUM_FEAT  512
#define HIDDEN    256
#define NUM_CLASS 64

// ---------- helpers ----------

static __device__ __forceinline__ bf16_t f2bf(float f) {
  unsigned u = __float_as_uint(f);
  unsigned r = (u + 0x7FFFu + ((u >> 16) & 1u)) >> 16;  // round-to-nearest-even
  unsigned short s = (unsigned short)r;
  return __builtin_bit_cast(bf16_t, s);
}

// A-fragment (16x32 bf16) per ISA 7.12.2: lane row = lane&15;
// elements 0..7  -> K = kk + base8 + e       (base8 = 8 for lanes 16..31)
// elements 8..15 -> K = kk + base8 + 16 + e
// p must already point at (row*stride + kk + base8)
static __device__ __forceinline__ v16bf load_a_frag(const bf16_t* p) {
  v8bf lo = *(const v8bf*)(p);
  v8bf hi = *(const v8bf*)(p + 16);
  v16bf a;
#pragma unroll
  for (int i = 0; i < 8; ++i) { a[i] = lo[i]; a[i + 8] = hi[i]; }
  return a;
}

// ---------- weight conversion: f32 row-major -> bf16 transposed ----------

__global__ void convw_kernel(const float* __restrict__ W1, const float* __restrict__ W2,
                             bf16_t* __restrict__ W1T, bf16_t* __restrict__ W2T) {
  int i = blockIdx.x * 256 + threadIdx.x;
  const int n1 = NUM_FEAT * HIDDEN;                 // 131072
  if (i < n1) {
    int f = i & (NUM_FEAT - 1);                     // K index
    int h = i >> 9;                                 // column
    W1T[i] = f2bf(W1[(size_t)f * HIDDEN + h]);      // W1T[h*512 + f]
  } else if (i < n1 + HIDDEN * NUM_CLASS) {
    int j = i - n1;
    int c = j >> 8;                                 // column
    int k = j & (HIDDEN - 1);                       // K index
    W2T[j] = f2bf(W2[(size_t)k * NUM_CLASS + c]);   // W2T[c*256 + k]
  }
}

// ---------- fused MLP: H = relu(X@W1+b1)@W2 + b2, 16 rows / block ----------

__global__ __launch_bounds__(128) void mlp_kernel(
    const float* __restrict__ X,
    const bf16_t* __restrict__ W1T, const float* __restrict__ bias1,
    const bf16_t* __restrict__ W2T, const float* __restrict__ bias2,
    float* __restrict__ H, int N) {
  __shared__ alignas(64) bf16_t Xs[16 * NUM_FEAT];   // 16 KB
  __shared__ alignas(64) bf16_t H1s[16 * HIDDEN];    //  8 KB

  const int tid  = threadIdx.x;
  const int lane = tid & 31;
  const int wave = tid >> 5;
  const int row0 = blockIdx.x * 16;

  // stage X tile as bf16 (coalesced f32 reads)
  for (int i = tid; i < 16 * NUM_FEAT; i += 128) {
    int r = i >> 9;
    int k = i & (NUM_FEAT - 1);
    int rr = row0 + r; if (rr >= N) rr = N - 1;
    Xs[i] = f2bf(X[(size_t)rr * NUM_FEAT + k]);
  }
  __syncthreads();

  const int m     = lane & 15;     // A row / B-D column within tile
  const int hi    = lane >> 4;
  const int base8 = hi << 3;
  const int khalf = hi << 4;

  // ---- GEMM1: each wave owns 4 of the 16 column tiles of H1 ----
  v8f zero = {0.f,0.f,0.f,0.f,0.f,0.f,0.f,0.f};
  v8f acc[4] = {zero, zero, zero, zero};
  const bf16_t* arow = Xs + m * NUM_FEAT + base8;
  const bf16_t* bp[4];
#pragma unroll
  for (int t = 0; t < 4; ++t)
    bp[t] = W1T + (size_t)((wave * 4 + t) * 16 + m) * NUM_FEAT + base8;

  for (int kk = 0; kk < NUM_FEAT; kk += 32) {
    v16bf a = load_a_frag(arow + kk);
#pragma unroll
    for (int t = 0; t < 4; ++t) {
      v16bf b = load_a_frag(bp[t] + kk);
      acc[t] = __builtin_amdgcn_wmma_f32_16x16x32_bf16(
          false, a, false, b, (short)0, acc[t], false, false);
    }
  }

  // bias + relu -> bf16 into LDS (layout: [row M][col], row stride HIDDEN)
#pragma unroll
  for (int t = 0; t < 4; ++t) {
    int col = (wave * 4 + t) * 16 + m;
    float bias = bias1[col];
#pragma unroll
    for (int v = 0; v < 8; ++v) {
      float x = acc[t][v] + bias;
      x = x > 0.f ? x : 0.f;
      H1s[(v + base8) * HIDDEN + col] = f2bf(x);
    }
  }
  __syncthreads();

  // ---- GEMM2: each wave owns one 16-col tile of the 64 output classes ----
  v8f c2 = zero;
  const bf16_t* a2  = H1s + m * HIDDEN + base8;
  const bf16_t* bp2 = W2T + (size_t)(wave * 16 + m) * HIDDEN + base8;
  for (int kk = 0; kk < HIDDEN; kk += 32) {
    v16bf a = load_a_frag(a2 + kk);
    v16bf b = load_a_frag(bp2 + kk);
    c2 = __builtin_amdgcn_wmma_f32_16x16x32_bf16(
        false, a, false, b, (short)0, c2, false, false);
  }
  int col = wave * 16 + m;
  float bias = bias2[col];
#pragma unroll
  for (int v = 0; v < 8; ++v) {
    int node = row0 + v + base8;
    if (node < N) H[(size_t)node * NUM_CLASS + col] = c2[v] + bias;
  }
}

// ---------- GCN normalization ----------

__global__ void fill1_kernel(float* __restrict__ p, int n) {
  int i = blockIdx.x * 256 + threadIdx.x;
  if (i < n) p[i] = 1.0f;                 // self-loop contributes 1 to degree
}

__global__ void deg_edges_kernel(const int* __restrict__ col, float* __restrict__ deg, int E) {
  int e = blockIdx.x * 256 + threadIdx.x;
  if (e < E) unsafeAtomicAdd(&deg[col[e]], 1.0f);
}

__global__ void rsqrt_kernel(float* __restrict__ p, int n) {
  int i = blockIdx.x * 256 + threadIdx.x;
  if (i < n) p[i] = rsqrtf(p[i]);         // deg >= 1 always (self-loops)
}

// ---------- GPR propagation ----------

__global__ void scale_init_kernel(float* __restrict__ out, const float* __restrict__ h,
                                  const float* __restrict__ temp, int NC) {
  int i = blockIdx.x * 256 + threadIdx.x;
  if (i < NC) out[i] = temp[0] * h[i];
}

// self-loop term: next[n][j] = dinv[n]^2 * cur[n][j]
__global__ void prop_self_kernel(float* __restrict__ nxt, const float* __restrict__ cur,
                                 const float* __restrict__ dinv, int NC) {
  int i = blockIdx.x * 256 + threadIdx.x;
  if (i < NC) {
    float d = dinv[i >> 6];
    nxt[i] = d * d * cur[i];
  }
}

// scatter-add over edges; one thread per (edge, 4-class chunk)
__global__ void prop_edges_kernel(const int* __restrict__ ei, const float* __restrict__ dinv,
                                  const float* __restrict__ cur, float* __restrict__ nxt, int E) {
  int gid = blockIdx.x * 256 + threadIdx.x;
  if (gid >= E * 16) return;
  int e  = gid >> 4;
  int cb = (gid & 15) << 2;
  int r = ei[e];
  int c = ei[E + e];
  float w = dinv[r] * dinv[c];
  const float4 v = *(const float4*)(cur + (size_t)r * NUM_CLASS + cb);
  float* dst = nxt + (size_t)c * NUM_CLASS + cb;
  unsafeAtomicAdd(dst + 0, w * v.x);
  unsafeAtomicAdd(dst + 1, w * v.y);
  unsafeAtomicAdd(dst + 2, w * v.z);
  unsafeAtomicAdd(dst + 3, w * v.w);
}

__global__ void accum_kernel(float* __restrict__ out, const float* __restrict__ nxt,
                             const float* __restrict__ temp, int kidx, int NC) {
  int i = blockIdx.x * 256 + threadIdx.x;
  if (i < NC) out[i] = fmaf(temp[kidx], nxt[i], out[i]);
}

// ---------- log_softmax: one wave32 per row of 64 classes ----------

__global__ __launch_bounds__(128) void logsoftmax_kernel(float* __restrict__ out, int N) {
  int lane = threadIdx.x & 31;
  int wave = threadIdx.x >> 5;
  int row  = blockIdx.x * 4 + wave;
  if (row >= N) return;
  float* p = out + (size_t)row * NUM_CLASS;
  float v0 = p[lane], v1 = p[lane + 32];
  float m = fmaxf(v0, v1);
#pragma unroll
  for (int off = 16; off >= 1; off >>= 1) m = fmaxf(m, __shfl_xor(m, off, 32));
  float s = __expf(v0 - m) + __expf(v1 - m);
#pragma unroll
  for (int off = 16; off >= 1; off >>= 1) s += __shfl_xor(s, off, 32);
  float l = m + __logf(s);
  p[lane]      = v0 - l;
  p[lane + 32] = v1 - l;
}

// ---------- host orchestration ----------

extern "C" void kernel_launch(void* const* d_in, const int* in_sizes, int n_in,
                              void* d_out, int out_size, void* d_ws, size_t ws_size,
                              hipStream_t stream) {
  const float* X    = (const float*)d_in[0];
  const int*   ei   = (const int*)d_in[1];
  const float* W1   = (const float*)d_in[2];
  const float* b1   = (const float*)d_in[3];
  const float* W2   = (const float*)d_in[4];
  const float* b2   = (const float*)d_in[5];
  const float* temp = (const float*)d_in[6];
  float* out = (float*)d_out;

  const int N = in_sizes[0] / NUM_FEAT;
  const int E = in_sizes[1] / 2;
  const int K = in_sizes[6] - 1;
  const int NC = N * NUM_CLASS;
  (void)n_in; (void)out_size; (void)ws_size;

  // workspace carve-out
  char* ws = (char*)d_ws;
  size_t off = 0;
  auto alloc = [&](size_t bytes) -> char* {
    char* p = ws + off;
    off += (bytes + 511) & ~(size_t)511;
    return p;
  };
  float*  dinv = (float*)alloc((size_t)N * sizeof(float));
  bf16_t* W1T  = (bf16_t*)alloc((size_t)NUM_FEAT * HIDDEN * sizeof(bf16_t));
  bf16_t* W2T  = (bf16_t*)alloc((size_t)HIDDEN * NUM_CLASS * sizeof(bf16_t));
  float*  bufA = (float*)alloc((size_t)NC * sizeof(float));
  float*  bufB = (float*)alloc((size_t)NC * sizeof(float));

  // 1) weights -> transposed bf16
  {
    int total = NUM_FEAT * HIDDEN + HIDDEN * NUM_CLASS;
    convw_kernel<<<(total + 255) / 256, 256, 0, stream>>>(W1, W2, W1T, W2T);
  }
  // 2) fused MLP (WMMA) -> bufA = h
  mlp_kernel<<<(N + 15) / 16, 128, 0, stream>>>(X, W1T, b1, W2T, b2, bufA, N);

  // 3) symmetric GCN normalization -> dinv
  fill1_kernel<<<(N + 255) / 256, 256, 0, stream>>>(dinv, N);
  deg_edges_kernel<<<(E + 255) / 256, 256, 0, stream>>>(ei + (size_t)E, dinv, E);
  rsqrt_kernel<<<(N + 255) / 256, 256, 0, stream>>>(dinv, N);

  // 4) hidden = temp[0] * h ; cur = h
  scale_init_kernel<<<(NC + 255) / 256, 256, 0, stream>>>(out, bufA, temp, NC);

  // 5) K propagation rounds
  float* cur = bufA;
  float* nxt = bufB;
  for (int k = 0; k < K; ++k) {
    prop_self_kernel<<<(NC + 255) / 256, 256, 0, stream>>>(nxt, cur, dinv, NC);
    {
      long long tot = (long long)E * 16;
      prop_edges_kernel<<<(int)((tot + 255) / 256), 256, 0, stream>>>(ei, dinv, cur, nxt, E);
    }
    accum_kernel<<<(NC + 255) / 256, 256, 0, stream>>>(out, nxt, temp, k + 1, NC);
    float* t = cur; cur = nxt; nxt = t;
  }

  // 6) row-wise log_softmax in place
  logsoftmax_kernel<<<(N + 3) / 4, 128, 0, stream>>>(out, N);
}
